// ImprovedSimVQQuantizer_87651692577319
// MI455X (gfx1250) — matrix operand order repaired and checked by reading, not compiled, not run on hardware
//
#include <hip/hip_runtime.h>
#include <hip/hip_bf16.h>

typedef __attribute__((ext_vector_type(16))) __bf16 v16bf;
typedef __attribute__((ext_vector_type(8)))  float  v8f;

#define D 256
#define KCODES 1024
#define STAGE 128           // codes staged in LDS per iteration
#define ROWP  272           // padded LDS row pitch in bf16 elems (544B = 32B-aligned, bank-skewed)
#define ROWS_PER_BLOCK 256  // 8 waves x 32 rows

// ---------------------------------------------------------------------------
// Kernel 0: zero counts + mse accumulator (ws is poisoned once by harness)
// ---------------------------------------------------------------------------
__global__ void vq_init(unsigned int* counts, unsigned long long* acc) {
    int t = blockIdx.x * blockDim.x + threadIdx.x;
    if (t < KCODES) counts[t] = 0u;
    if (t == 0) *acc = 0ull;
}

// ---------------------------------------------------------------------------
// Kernel 1: codebook = normalize(latent_basis @ W^T) * sqrt(D)
// Also emits bf16 hi/lo split and per-row ||c||^2.
// grid = K blocks, 256 threads (thread i -> output dim i of code k)
// ---------------------------------------------------------------------------
__global__ __launch_bounds__(256) void vq_codebook(
    const float* __restrict__ basis, const float* __restrict__ W,
    float* __restrict__ cb, float* __restrict__ cnorm2,
    __bf16* __restrict__ cb_hi, __bf16* __restrict__ cb_lo)
{
    const int k = blockIdx.x;
    const int i = threadIdx.x;
    const float* brow = basis + (size_t)k * D;
    const float* wrow = W + (size_t)i * D;   // cb[k][i] = sum_j basis[k][j] * W[i][j]
    float dot = 0.f;
#pragma unroll 8
    for (int j = 0; j < D; ++j) dot = fmaf(brow[j], wrow[j], dot);

    __shared__ float red[256];
    red[i] = dot * dot;
    __syncthreads();
    for (int s = 128; s > 0; s >>= 1) {
        if (i < s) red[i] += red[i + s];
        __syncthreads();
    }
    const float norm  = sqrtf(red[0]);
    const float scale = 16.0f / fmaxf(norm, 1e-12f);   // sqrt(256) = 16
    const float c = dot * scale;
    cb[(size_t)k * D + i] = c;

    __bf16 h = (__bf16)c;
    cb_hi[(size_t)k * D + i] = h;
    cb_lo[(size_t)k * D + i] = (__bf16)(c - (float)h);

    __syncthreads();
    red[i] = c * c;
    __syncthreads();
    for (int s = 128; s > 0; s >>= 1) {
        if (i < s) red[i] += red[i + s];
        __syncthreads();
    }
    if (i == 0) cnorm2[k] = red[0];
}

// ---------------------------------------------------------------------------
// Kernel 2: main fused GEMM + argmin + gather + mse + histogram.
// 256 threads = 8 waves; wave owns 32 rows (2 register-resident A tiles);
// block owns 256 rows. Each staged B chunk feeds 6 WMMAs (2 tiles x 3-term
// bf16 split: hi*hi + lo*hi + hi*lo) with f32 accumulation.
// ---------------------------------------------------------------------------
__global__ __launch_bounds__(256) void vq_main(
    const float* __restrict__ z,
    const float* __restrict__ cb,
    const __bf16* __restrict__ cb_hi, const __bf16* __restrict__ cb_lo,
    const float* __restrict__ cnorm2,
    float* __restrict__ out_zq, float* __restrict__ out_idx,
    unsigned int* __restrict__ counts, unsigned long long* __restrict__ mse_acc)
{
    __shared__ __align__(32) __bf16 sB_hi[STAGE * ROWP];
    __shared__ __align__(32) __bf16 sB_lo[STAGE * ROWP];
    __shared__ float sCn[STAGE];
    __shared__ int   sWin[ROWS_PER_BLOCK];
    __shared__ float sRed[256];

    const int tid   = threadIdx.x;
    const int wave  = tid >> 5;
    const int lane  = tid & 31;
    const int l16   = lane & 15;
    const bool hiHalf = (lane >= 16);
    const int blockRow0 = blockIdx.x * ROWS_PER_BLOCK;
    const int waveRow0  = blockRow0 + wave * 32;

    // ---- A tiles: 2 x (16 rows x 256 dims), bf16 hi/lo in WMMA A layout.
    // A k order used here is contiguous: lanes 0-15 -> M=lane, K elems {0..15};
    //                 lanes 16-31 -> M=lane-16, K elems {16..31}.
    v16bf a_hi0[8], a_lo0[8], a_hi1[8], a_lo1[8];
#pragma unroll
    for (int t = 0; t < 2; ++t) {
        const int row = waveRow0 + t * 16 + l16;
        const float* zr = z + (size_t)row * D;
#pragma unroll
        for (int c = 0; c < 8; ++c) {
            const int s0 = c * 32 + (hiHalf ? 16 : 0);
            float4 f0 = *(const float4*)(zr + s0);
            float4 f1 = *(const float4*)(zr + s0 + 4);
            float4 f2 = *(const float4*)(zr + s0 + 8);
            float4 f3 = *(const float4*)(zr + s0 + 12);
            float v[16] = { f0.x,f0.y,f0.z,f0.w, f1.x,f1.y,f1.z,f1.w,
                            f2.x,f2.y,f2.z,f2.w, f3.x,f3.y,f3.z,f3.w };
            v16bf h, l;
#pragma unroll
            for (int e = 0; e < 16; ++e) {
                __bf16 hb = (__bf16)v[e];
                h[e] = hb;
                l[e] = (__bf16)(v[e] - (float)hb);
            }
            if (t == 0) { a_hi0[c] = h; a_lo0[c] = l; }
            else        { a_hi1[c] = h; a_lo1[c] = l; }
        }
    }

    float best0[8], best1[8];
    int   bidx0[8], bidx1[8];
#pragma unroll
    for (int r = 0; r < 8; ++r) {
        best0[r] = 3.0e38f; bidx0[r] = 0;
        best1[r] = 3.0e38f; bidx1[r] = 0;
    }

    for (int stg = 0; stg < KCODES / STAGE; ++stg) {
        const int k0 = stg * STAGE;
        __syncthreads();
        // cooperative stage of codebook hi/lo into padded LDS rows
        {
            const uint4* ghi = (const uint4*)(cb_hi + (size_t)k0 * D);
            const uint4* glo = (const uint4*)(cb_lo + (size_t)k0 * D);
            uint4* shi = (uint4*)sB_hi;
            uint4* slo = (uint4*)sB_lo;
            // STAGE rows * 32 uint4 per row (512B of data, pitch 34 uint4)
#pragma unroll
            for (int t = 0; t < (STAGE * 32) / 256; ++t) {
                const int li  = t * 256 + tid;
                const int r   = li >> 5;
                const int c16 = li & 31;
                shi[r * 34 + c16] = ghi[r * 32 + c16];
                slo[r * 34 + c16] = glo[r * 32 + c16];
            }
            if (tid < STAGE) sCn[tid] = cnorm2[k0 + tid];
            if (stg + 1 < KCODES / STAGE) {
                __builtin_prefetch(cb_hi + (size_t)(k0 + STAGE) * D + tid * 64, 0, 1);
                __builtin_prefetch(cb_lo + (size_t)(k0 + STAGE) * D + tid * 64, 0, 1);
            }
        }
        __syncthreads();

        // each wave sweeps the 16-code tiles of this stage
#pragma unroll 1
        for (int nt = 0; nt < STAGE / 16; ++nt) {
            const int nbase = nt * 16;
            v8f acc0 = {};
            v8f acc1 = {};
#pragma unroll
            for (int c = 0; c < 8; ++c) {
                // B uses the same contiguous k order as A: lane = N column (l16); lanes 0-15 hold K 0..15,
                // lanes 16-31 hold K 16..31 of the 32-chunk.
                const int koff = c * 32 + (hiHalf ? 16 : 0);
                const v16bf bh = *(const v16bf*)(sB_hi + (size_t)(nbase + l16) * ROWP + koff);
                const v16bf bl = *(const v16bf*)(sB_lo + (size_t)(nbase + l16) * ROWP + koff);
                acc0 = __builtin_amdgcn_wmma_f32_16x16x32_bf16(false, a_hi0[c], false, bh, (short)0, acc0, false, false);
                acc1 = __builtin_amdgcn_wmma_f32_16x16x32_bf16(false, a_hi1[c], false, bh, (short)0, acc1, false, false);
                acc0 = __builtin_amdgcn_wmma_f32_16x16x32_bf16(false, a_lo0[c], false, bh, (short)0, acc0, false, false);
                acc1 = __builtin_amdgcn_wmma_f32_16x16x32_bf16(false, a_lo1[c], false, bh, (short)0, acc1, false, false);
                acc0 = __builtin_amdgcn_wmma_f32_16x16x32_bf16(false, a_hi0[c], false, bl, (short)0, acc0, false, false);
                acc1 = __builtin_amdgcn_wmma_f32_16x16x32_bf16(false, a_hi1[c], false, bl, (short)0, acc1, false, false);
            }
            const int   n  = k0 + nbase + l16;
            const float cn = sCn[nbase + l16];
#pragma unroll
            for (int r = 0; r < 8; ++r) {
                const float s0 = cn - 2.0f * acc0[r];   // ||c||^2 - 2 z.c
                const float s1 = cn - 2.0f * acc1[r];
                if (s0 < best0[r]) { best0[r] = s0; bidx0[r] = n; }  // ascending n => ties keep low idx
                if (s1 < best1[r]) { best1[r] = s1; bidx1[r] = n; }
            }
        }
    }

    // cross-lane argmin within each 16-lane half (C/D layout: lanes0-15 -> M=r, lanes16-31 -> M=r+8)
#pragma unroll
    for (int r = 0; r < 8; ++r) {
#pragma unroll
        for (int m = 1; m < 16; m <<= 1) {
            float ov = __shfl_xor(best0[r], m, 32);
            int   oi = __shfl_xor(bidx0[r], m, 32);
            if (ov < best0[r] || (ov == best0[r] && oi < bidx0[r])) { best0[r] = ov; bidx0[r] = oi; }
            ov = __shfl_xor(best1[r], m, 32);
            oi = __shfl_xor(bidx1[r], m, 32);
            if (ov < best1[r] || (ov == best1[r] && oi < bidx1[r])) { best1[r] = ov; bidx1[r] = oi; }
        }
    }
    if (lane == 0 || lane == 16) {
        const int mb0 = wave * 32 + (hiHalf ? 8 : 0);
#pragma unroll
        for (int r = 0; r < 8; ++r) {
            sWin[mb0 + r]      = bidx0[r];
            sWin[mb0 + 16 + r] = bidx1[r];
        }
    }
    __syncthreads();

    // ---- gather z_q, write outputs, accumulate mse, histogram
    float part = 0.f;
    for (int m = 0; m < ROWS_PER_BLOCK; ++m) {
        const int row = blockRow0 + m;
        const int k   = sWin[m];
        const float ze   = z [(size_t)row * D + tid];
        const float cv   = cb[(size_t)k   * D + tid];
        const float diff = cv - ze;                   // stopgrad(z_q - z_e)
        out_zq[(size_t)row * D + tid] = ze + diff;    // straight-through (mimics ref rounding)
        part += diff * diff;
    }
    {
        const int row = blockRow0 + tid;              // 256 threads cover 256 rows
        const int k   = sWin[tid];
        out_idx[row] = (float)k;
        atomicAdd(&counts[k], 1u);
    }
    sRed[tid] = part;
    __syncthreads();
    for (int s = 128; s > 0; s >>= 1) {
        if (tid < s) sRed[tid] += sRed[tid + s];
        __syncthreads();
    }
    if (tid == 0) {
        // fixed-point commit (order-independent => deterministic across replays)
        const double scaled = (double)sRed[0] * 1048576.0;
        atomicAdd(mse_acc, (unsigned long long)(scaled + 0.5));
    }
}

// ---------------------------------------------------------------------------
// Kernel 3: scalars — vq_loss, perplexity, codebook usage
// ---------------------------------------------------------------------------
__global__ __launch_bounds__(256) void vq_final(
    const unsigned int* __restrict__ counts,
    const unsigned long long* __restrict__ mse_acc,
    float* out_loss, float* out_ppl, float* out_usage, int nRows)
{
    __shared__ float sH[256];
    __shared__ unsigned int sU[256];
    const int tid = threadIdx.x;
    float h = 0.f; unsigned u = 0;
    const float invN = 1.0f / (float)nRows;
    for (int t = tid; t < KCODES; t += 256) {
        const unsigned c = counts[t];
        const float p = (float)c * invN;
        h += p * logf(p + 1e-10f);
        u += (c > 0u) ? 1u : 0u;
    }
    sH[tid] = h; sU[tid] = u;
    __syncthreads();
    for (int s = 128; s > 0; s >>= 1) {
        if (tid < s) { sH[tid] += sH[tid + s]; sU[tid] += sU[tid + s]; }
        __syncthreads();
    }
    if (tid == 0) {
        const double mse = ((double)*mse_acc / 1048576.0) / ((double)nRows * (double)D);
        const float m = (float)mse;
        *out_loss  = m + 0.25f * m;        // codebook + commitment
        *out_ppl   = expf(-sH[0]);
        *out_usage = (float)sU[0] / (float)KCODES;
    }
}

// ---------------------------------------------------------------------------
extern "C" void kernel_launch(void* const* d_in, const int* in_sizes, int n_in,
                              void* d_out, int out_size, void* d_ws, size_t ws_size,
                              hipStream_t stream) {
    const float* z_e   = (const float*)d_in[0];   // [N, 256]
    const float* basis = (const float*)d_in[1];   // [1024, 256]
    const float* W     = (const float*)d_in[2];   // [256, 256]

    const int N = in_sizes[0] / D;                // 65536 rows

    // workspace carve-up (all offsets 8B-aligned)
    char* ws = (char*)d_ws;
    float*              cb     = (float*)(ws + 0);                     // 1 MB
    float*              cnorm2 = (float*)(ws + 1048576);               // 4 KB
    __bf16*             cb_hi  = (__bf16*)(ws + 1052672);              // 512 KB
    __bf16*             cb_lo  = (__bf16*)(ws + 1576960);              // 512 KB
    unsigned int*       counts = (unsigned int*)(ws + 2101248);        // 4 KB
    unsigned long long* acc    = (unsigned long long*)(ws + 2105344);  // 8 B

    // output layout: [z_q_ste N*256][vq_loss][indices N][perplexity][usage]
    float* out_zq    = (float*)d_out;
    float* out_loss  = out_zq + (size_t)N * D;
    float* out_idx   = out_loss + 1;
    float* out_ppl   = out_idx + N;
    float* out_usage = out_ppl + 1;

    vq_init<<<4, 256, 0, stream>>>(counts, acc);
    vq_codebook<<<KCODES, 256, 0, stream>>>(basis, W, cb, cnorm2, cb_hi, cb_lo);
    vq_main<<<N / ROWS_PER_BLOCK, 256, 0, stream>>>(z_e, cb, cb_hi, cb_lo, cnorm2,
                                                    out_zq, out_idx, counts, acc);
    vq_final<<<1, 256, 0, stream>>>(counts, acc, out_loss, out_ppl, out_usage, N);
}
